// VisualDecoderLayer_80144089743526
// MI455X (gfx1250) — matrix operand order repaired and checked by reading, not compiled, not run on hardware
//
#include <hip/hip_runtime.h>
#include <hip/hip_bf16.h>
#include <math.h>

typedef __attribute__((ext_vector_type(16))) _Float16 v16h;
typedef __attribute__((ext_vector_type(2)))  _Float16 h2v;
typedef __attribute__((ext_vector_type(8)))  float    v8f;
typedef __attribute__((ext_vector_type(4)))  float    f4v;

#define D_MODEL 256
#define NQ      1024
#define BATCH   4
#define LV      16384           // 4*64*64
#define CA_INNER 512
#define M_Q     (BATCH*NQ)      // 4096
#define M_KV    (BATCH*LV)      // 65536

static __device__ __forceinline__ int imin(int a, int b) { return a < b ? a : b; }

// ---------------------------------------------------------------------------
// Weight convert+transpose: W [K,N] fp32 -> Wt [N,K] f16
// ---------------------------------------------------------------------------
__global__ void wt_f16_kernel(const float* __restrict__ W, _Float16* __restrict__ Wt,
                              int K, int N) {
    int idx = blockIdx.x * blockDim.x + threadIdx.x;
    if (idx >= K * N) return;
    int n = idx / K, k = idx - n * K;           // contiguous writes over k
    Wt[idx] = (_Float16)W[(size_t)k * N + n];
}

// ---------------------------------------------------------------------------
// WMMA GEMM:  C[M,N] = act( (A0 (+A1)) @ W + bias ),  Wt = W^T as [N,K] f16
// A fp32 row-major, f16 conversion on the fly. Each wave: 32x16 output
// (two M-tiles sharing one B fragment -> 2 v_wmma per K-step), 4 waves
// (4 N-tiles) per block share A through the cache. HAS_A1 is a compile-time
// flag so the fused residual add has no per-lane pointer test.
// ---------------------------------------------------------------------------
template <bool HAS_A1>
__global__ __launch_bounds__(128)
void gemm_wmma_kernel(const float* __restrict__ A0, const float* __restrict__ A1,
                      const _Float16* __restrict__ Wt, const float* __restrict__ bias,
                      float* __restrict__ Cf, _Float16* __restrict__ Ch,
                      int M, int N, int K, int act) {
    const int lane = threadIdx.x & 31;
    const int wave = threadIdx.x >> 5;
    const int ntile = blockIdx.x * 4 + wave;
    if (ntile * 16 >= N) return;                // wave-uniform exit
    const int l16 = lane & 15, hs = lane >> 4;
    const int m0 = blockIdx.y * 32, n0 = ntile * 16;

    const float* a0p0 = A0 + (size_t)(m0 + l16) * K;          // M-tile 0 row
    const float* a0p1 = a0p0 + (size_t)16 * K;                // M-tile 1 row
    const float* a1p0 = HAS_A1 ? (A1 + (size_t)(m0 + l16) * K) : nullptr;
    const float* a1p1 = HAS_A1 ? (a1p0 + (size_t)16 * K) : nullptr;
    const _Float16* bp = Wt + (size_t)(n0 + l16) * K + hs * 8;

    v8f acc0 = {}, acc1 = {};
    for (int kk = 0; kk < K; kk += 32) {
        __builtin_prefetch(a0p0 + kk + 128, 0, 3);  // global_prefetch_b8 (near)
        __builtin_prefetch(bp + kk + 256, 0, 3);
        const int ka = kk + hs * 8;
        // A fragments: rows m0+l16 (+16), K runs [ka,+8) and [ka+16,+8)
        f4v q00 = *(const f4v*)(a0p0 + ka);
        f4v q01 = *(const f4v*)(a0p0 + ka + 4);
        f4v q02 = *(const f4v*)(a0p0 + ka + 16);
        f4v q03 = *(const f4v*)(a0p0 + ka + 20);
        f4v q10 = *(const f4v*)(a0p1 + ka);
        f4v q11 = *(const f4v*)(a0p1 + ka + 4);
        f4v q12 = *(const f4v*)(a0p1 + ka + 16);
        f4v q13 = *(const f4v*)(a0p1 + ka + 20);
        if (HAS_A1) {
            q00 += *(const f4v*)(a1p0 + ka);
            q01 += *(const f4v*)(a1p0 + ka + 4);
            q02 += *(const f4v*)(a1p0 + ka + 16);
            q03 += *(const f4v*)(a1p0 + ka + 20);
            q10 += *(const f4v*)(a1p1 + ka);
            q11 += *(const f4v*)(a1p1 + ka + 4);
            q12 += *(const f4v*)(a1p1 + ka + 16);
            q13 += *(const f4v*)(a1p1 + ka + 20);
        }
        v16h a0, a1;
        #pragma unroll
        for (int e = 0; e < 4; ++e) {
            a0[e]      = (_Float16)q00[e];  a0[e + 4]  = (_Float16)q01[e];
            a0[e + 8]  = (_Float16)q02[e];  a0[e + 12] = (_Float16)q03[e];
            a1[e]      = (_Float16)q10[e];  a1[e + 4]  = (_Float16)q11[e];
            a1[e + 8]  = (_Float16)q12[e];  a1[e + 12] = (_Float16)q13[e];
        }
        // B fragment: col = n0+l16, K runs [kk+hs*8,+8) and [kk+16+hs*8,+8)
        v16h b;
        #pragma unroll
        for (int e = 0; e < 8; ++e) { b[e] = bp[kk + e]; b[e + 8] = bp[kk + 16 + e]; }
        acc0 = __builtin_amdgcn_wmma_f32_16x16x32_f16(false, a0, false, b,
                                                      (short)0, acc0, false, false);
        acc1 = __builtin_amdgcn_wmma_f32_16x16x32_f16(false, a1, false, b,
                                                      (short)0, acc1, false, false);
    }
    const int cn = n0 + l16;
    const float bv = bias ? bias[cn] : 0.0f;
    #pragma unroll
    for (int r = 0; r < 8; ++r) {
        const int cm = m0 + r + hs * 8;
        float v0 = acc0[r] + bv;
        float v1 = acc1[r] + bv;
        if (act == 1) { v0 = fmaxf(v0, 0.0f); v1 = fmaxf(v1, 0.0f); }
        size_t o0 = (size_t)cm * N + cn;
        size_t o1 = (size_t)(cm + 16) * N + cn;
        if (Cf) { Cf[o0] = v0; Cf[o1] = v1; }
        if (Ch) { Ch[o0] = (_Float16)v0; Ch[o1] = (_Float16)v1; }
    }
}

// ---------------------------------------------------------------------------
// Flash self-attention: one wave per (batch, head, 16-query tile).
// dh = 32 (one WMMA K-step), online softmax over 1024 keys in chunks of 32.
// ---------------------------------------------------------------------------
__global__ __launch_bounds__(32)
void flash_attn_kernel(const float* __restrict__ Q, const float* __restrict__ K,
                       const float* __restrict__ V, float* __restrict__ O) {
    __shared__ _Float16 Pl[16][32];
    __shared__ float    Vl[32][36];             // stride 36 floats keeps 16B align
    const int lane = threadIdx.x;
    const int qt = blockIdx.x, h = blockIdx.y, b = blockIdx.z;
    const int l16 = lane & 15, hs = lane >> 4;
    const float scale = 0.17677669529663687f;   // 1/sqrt(32)

    // Q fragment (A layout), pre-scaled
    const float* qp = Q + ((size_t)(b * NQ + qt * 16 + l16)) * D_MODEL + h * 32;
    v16h aq;
    {
        f4v t0 = *(const f4v*)(qp + hs * 8);
        f4v t1 = *(const f4v*)(qp + hs * 8 + 4);
        f4v t2 = *(const f4v*)(qp + 16 + hs * 8);
        f4v t3 = *(const f4v*)(qp + 16 + hs * 8 + 4);
        #pragma unroll
        for (int e = 0; e < 4; ++e) {
            aq[e]      = (_Float16)(t0[e] * scale);
            aq[e + 4]  = (_Float16)(t1[e] * scale);
            aq[e + 8]  = (_Float16)(t2[e] * scale);
            aq[e + 12] = (_Float16)(t3[e] * scale);
        }
    }

    v8f o0 = {}, o1 = {};
    float mrun[8], lrun[8];
    #pragma unroll
    for (int r = 0; r < 8; ++r) { mrun[r] = -1e30f; lrun[r] = 0.0f; }

    for (int kk = 0; kk < NQ; kk += 32) {
        // K fragments (B layout): columns = keys, K-dim = dh
        const float* kp0 = K + ((size_t)(b * NQ + kk + l16)) * D_MODEL + h * 32 + hs * 8;
        const float* kp1 = kp0 + (size_t)16 * D_MODEL;
        v16h bk0, bk1;
        {
            f4v u0 = *(const f4v*)(kp0),      u1 = *(const f4v*)(kp0 + 4);
            f4v u2 = *(const f4v*)(kp0 + 16), u3 = *(const f4v*)(kp0 + 20);
            f4v w0 = *(const f4v*)(kp1),      w1 = *(const f4v*)(kp1 + 4);
            f4v w2 = *(const f4v*)(kp1 + 16), w3 = *(const f4v*)(kp1 + 20);
            #pragma unroll
            for (int e = 0; e < 4; ++e) {
                bk0[e]      = (_Float16)u0[e];  bk0[e + 4]  = (_Float16)u1[e];
                bk0[e + 8]  = (_Float16)u2[e];  bk0[e + 12] = (_Float16)u3[e];
                bk1[e]      = (_Float16)w0[e];  bk1[e + 4]  = (_Float16)w1[e];
                bk1[e + 8]  = (_Float16)w2[e];  bk1[e + 12] = (_Float16)w3[e];
            }
        }
        v8f s0 = {}, s1 = {};
        s0 = __builtin_amdgcn_wmma_f32_16x16x32_f16(false, aq, false, bk0, (short)0, s0, false, false);
        s1 = __builtin_amdgcn_wmma_f32_16x16x32_f16(false, aq, false, bk1, (short)0, s1, false, false);

        // stage V chunk [32 keys x 32 dh] into LDS (vectorized b128)
        const float* vp = V + ((size_t)(b * NQ + kk + lane)) * D_MODEL + h * 32;
        #pragma unroll
        for (int j = 0; j < 8; ++j)
            *(f4v*)(&Vl[lane][j * 4]) = *(const f4v*)(vp + j * 4);

        // online softmax (per row r -> global row r + hs*8)
        #pragma unroll
        for (int r = 0; r < 8; ++r) {
            float mx = fmaxf(s0[r], s1[r]);
            #pragma unroll
            for (int m = 8; m >= 1; m >>= 1) mx = fmaxf(mx, __shfl_xor(mx, m, 32));
            float mnew = fmaxf(mrun[r], mx);
            float p0 = __expf(s0[r] - mnew), p1 = __expf(s1[r] - mnew);
            s0[r] = p0; s1[r] = p1;
            float sm = p0 + p1;
            #pragma unroll
            for (int m = 8; m >= 1; m >>= 1) sm += __shfl_xor(sm, m, 32);
            float f = __expf(mrun[r] - mnew);
            lrun[r] = lrun[r] * f + sm;
            mrun[r] = mnew;
            o0[r] *= f; o1[r] *= f;
        }
        // P: C-layout -> LDS -> A-layout
        #pragma unroll
        for (int r = 0; r < 8; ++r) {
            Pl[r + hs * 8][l16]      = (_Float16)s0[r];
            Pl[r + hs * 8][16 + l16] = (_Float16)s1[r];
        }
        __syncthreads();
        v16h ap;
        #pragma unroll
        for (int e = 0; e < 8; ++e) {
            ap[e]     = Pl[l16][hs * 16 + e];
            ap[e + 8] = Pl[l16][hs * 16 + 8 + e];
        }
        v16h bv0, bv1;
        #pragma unroll
        for (int e = 0; e < 16; ++e) {
            bv0[e] = (_Float16)Vl[hs * 16 + e][l16];
            bv1[e] = (_Float16)Vl[hs * 16 + e][16 + l16];
        }
        o0 = __builtin_amdgcn_wmma_f32_16x16x32_f16(false, ap, false, bv0, (short)0, o0, false, false);
        o1 = __builtin_amdgcn_wmma_f32_16x16x32_f16(false, ap, false, bv1, (short)0, o1, false, false);
        __syncthreads();
    }
    float* op = O + ((size_t)(b * NQ + qt * 16)) * D_MODEL + h * 32;
    #pragma unroll
    for (int r = 0; r < 8; ++r) {
        float inv = 1.0f / lrun[r];
        op[(size_t)(r + hs * 8) * D_MODEL + l16]      = o0[r] * inv;
        op[(size_t)(r + hs * 8) * D_MODEL + 16 + l16] = o1[r] * inv;
    }
}

// ---------------------------------------------------------------------------
// Fused residual add + LayerNorm: out = LN(A + B) * g + be ; one row per block
// ---------------------------------------------------------------------------
__global__ __launch_bounds__(256)
void add_ln_kernel(const float* __restrict__ A, const float* __restrict__ Bm,
                   const float* __restrict__ g, const float* __restrict__ be,
                   float* __restrict__ out) {
    __shared__ float red[256];
    const int row = blockIdx.x, t = threadIdx.x;
    const size_t idx = (size_t)row * D_MODEL + t;
    float v = A[idx] + Bm[idx];
    red[t] = v; __syncthreads();
    for (int s = 128; s > 0; s >>= 1) { if (t < s) red[t] += red[t + s]; __syncthreads(); }
    float mean = red[0] * (1.0f / D_MODEL);
    __syncthreads();
    float d = v - mean;
    red[t] = d * d; __syncthreads();
    for (int s = 128; s > 0; s >>= 1) { if (t < s) red[t] += red[t + s]; __syncthreads(); }
    float var = red[0] * (1.0f / D_MODEL);
    out[idx] = d * rsqrtf(var + 1e-5f) * g[t] + be[t];
}

// ---------------------------------------------------------------------------
// Reference points: ref = sigmoid(x @ Wref + bref), N=3 (too small for WMMA)
// ---------------------------------------------------------------------------
__global__ void ref_kernel(const float* __restrict__ x, const float* __restrict__ W,
                           const float* __restrict__ bias, float* __restrict__ ref) {
    int i = blockIdx.x * blockDim.x + threadIdx.x;
    if (i >= M_Q * 3) return;
    int m = i / 3, j = i - m * 3;
    float s = bias[j];
    const float* xp = x + (size_t)m * D_MODEL;
    for (int k = 0; k < D_MODEL; ++k) s += xp[k] * W[k * 3 + j];
    ref[i] = 1.0f / (1.0f + __expf(-s));
}

// ---------------------------------------------------------------------------
// Deformable trilinear sampling + joint softmax attention over 2*4 points.
// One block per (b,n), one wave per head; lane handles 2 of 64 dh dims.
// ---------------------------------------------------------------------------
__global__ __launch_bounds__(256)
void deform_sample_kernel(const float* __restrict__ qh, const float* __restrict__ ref,
                          const float* __restrict__ off,
                          const _Float16* __restrict__ kvol, const _Float16* __restrict__ vvol,
                          const _Float16* __restrict__ fkvol, const _Float16* __restrict__ fvvol,
                          float* __restrict__ samp) {
    const int bn = blockIdx.x;                  // 0..4095
    const int h = threadIdx.x >> 5;
    const int lane = threadIdx.x & 31;
    const int b = bn >> 10;
    const size_t qbase = (size_t)bn * CA_INNER + h * 64 + lane * 2;
    const float q0 = qh[qbase], q1 = qh[qbase + 1];
    const float r0 = ref[bn * 3 + 0], r1 = ref[bn * 3 + 1], r2 = ref[bn * 3 + 2];

    float lk[4], lf[4], vs0[4], vs1[4], fv0[4], fv1[4];
    #pragma unroll
    for (int p = 0; p < 4; ++p) {
        const int ob = bn * 96 + (h * 4 + p) * 3;
        float t  = r0 + off[ob + 0] * (2.0f / 4.0f);
        float y  = r1 + off[ob + 1] * (2.0f / 64.0f);
        float xx = r2 + off[ob + 2] * (2.0f / 64.0f);
        t  = fminf(fmaxf(t, 0.0f), 1.0f) * 3.0f;
        y  = fminf(fmaxf(y, 0.0f), 1.0f) * 63.0f;
        xx = fminf(fmaxf(xx, 0.0f), 1.0f) * 63.0f;
        const int t0 = (int)floorf(t), y0 = (int)floorf(y), x0 = (int)floorf(xx);
        const float ft = t - (float)t0, fy = y - (float)y0, fx = xx - (float)x0;
        float dk = 0.f, df = 0.f, a0 = 0.f, a1 = 0.f, c0 = 0.f, c1 = 0.f;
        #pragma unroll
        for (int c = 0; c < 8; ++c) {
            const int dt = c >> 2, dy = (c >> 1) & 1, dx = c & 1;
            const int ct = imin(t0 + dt, 3), cy = imin(y0 + dy, 63), cx = imin(x0 + dx, 63);
            const float w = (dt ? ft : 1.0f - ft) * (dy ? fy : 1.0f - fy) * (dx ? fx : 1.0f - fx);
            const size_t idx = ((size_t)b * LV + (size_t)(ct * 4096 + cy * 64 + cx)) * CA_INNER
                             + h * 64 + lane * 2;
            h2v kv = *(const h2v*)(kvol + idx);
            h2v fk = *(const h2v*)(fkvol + idx);
            h2v vv = *(const h2v*)(vvol + idx);
            h2v fv = *(const h2v*)(fvvol + idx);
            dk += w * ((float)kv.x * q0 + (float)kv.y * q1);
            df += w * ((float)fk.x * q0 + (float)fk.y * q1);
            a0 += w * (float)vv.x; a1 += w * (float)vv.y;
            c0 += w * (float)fv.x; c1 += w * (float)fv.y;
        }
        #pragma unroll
        for (int m = 16; m >= 1; m >>= 1) { dk += __shfl_xor(dk, m, 32); df += __shfl_xor(df, m, 32); }
        lk[p] = dk * 0.125f; lf[p] = df * 0.125f;   // 1/sqrt(64)
        vs0[p] = a0; vs1[p] = a1; fv0[p] = c0; fv1[p] = c1;
    }
    float mx = -1e30f;
    #pragma unroll
    for (int p = 0; p < 4; ++p) { mx = fmaxf(mx, lk[p]); mx = fmaxf(mx, lf[p]); }
    float se = 0.f, ek[4], ef[4];
    #pragma unroll
    for (int p = 0; p < 4; ++p) {
        ek[p] = __expf(lk[p] - mx); ef[p] = __expf(lf[p] - mx);
        se += ek[p] + ef[p];
    }
    const float inv = 1.0f / se;
    float o0 = 0.f, o1 = 0.f;
    #pragma unroll
    for (int p = 0; p < 4; ++p) {
        o0 += ek[p] * vs0[p] + ef[p] * fv0[p];
        o1 += ek[p] * vs1[p] + ef[p] * fv1[p];
    }
    samp[qbase]     = o0 * inv;
    samp[qbase + 1] = o1 * inv;
}

// ---------------------------------------------------------------------------
// Host orchestration
// ---------------------------------------------------------------------------
extern "C" void kernel_launch(void* const* d_in, const int* in_sizes, int n_in,
                              void* d_out, int out_size, void* d_ws, size_t ws_size,
                              hipStream_t stream) {
    (void)in_sizes; (void)n_in; (void)out_size; (void)ws_size;
    const float* query  = (const float*)d_in[0];
    const float* qpos   = (const float*)d_in[1];
    const float* img    = (const float*)d_in[2];
    const float* flow   = (const float*)d_in[3];
    const float* pos    = (const float*)d_in[4];
    const float* sa_Wq = (const float*)d_in[6];  const float* sa_Wk = (const float*)d_in[7];
    const float* sa_Wv = (const float*)d_in[8];  const float* sa_Wo = (const float*)d_in[9];
    const float* sa_bq = (const float*)d_in[10]; const float* sa_bk = (const float*)d_in[11];
    const float* sa_bv = (const float*)d_in[12]; const float* sa_bo = (const float*)d_in[13];
    const float* ln1_b = (const float*)d_in[14]; const float* ln2_b = (const float*)d_in[15];
    const float* ln3_b = (const float*)d_in[16];
    const float* ln1_g = (const float*)d_in[17]; const float* ln2_g = (const float*)d_in[18];
    const float* ln3_g = (const float*)d_in[19];
    const float* ca_Wq = (const float*)d_in[20]; const float* ca_bq = (const float*)d_in[21];
    const float* ca_Wk = (const float*)d_in[22]; const float* ca_bk = (const float*)d_in[23];
    const float* ca_Wv = (const float*)d_in[24]; const float* ca_bv = (const float*)d_in[25];
    const float* ca_Wref = (const float*)d_in[26]; const float* ca_bref = (const float*)d_in[27];
    const float* ca_Woff = (const float*)d_in[28]; const float* ca_boff = (const float*)d_in[29];
    const float* ca_Wo = (const float*)d_in[30]; const float* ca_bo = (const float*)d_in[31];
    const float* f_W1 = (const float*)d_in[32];  const float* f_b1 = (const float*)d_in[33];
    const float* f_W2 = (const float*)d_in[34];  const float* f_b2 = (const float*)d_in[35];

    char* ws = (char*)d_ws;
    size_t wso = 0;
    auto alloc = [&](size_t b) -> void* {
        void* p = ws + wso; wso += (b + 255) & ~(size_t)255; return p;
    };
    _Float16* WtSaQ  = (_Float16*)alloc((size_t)256 * 256 * 2);
    _Float16* WtSaK  = (_Float16*)alloc((size_t)256 * 256 * 2);
    _Float16* WtSaV  = (_Float16*)alloc((size_t)256 * 256 * 2);
    _Float16* WtSaO  = (_Float16*)alloc((size_t)256 * 256 * 2);
    _Float16* WtCaQ  = (_Float16*)alloc((size_t)512 * 256 * 2);
    _Float16* WtCaK  = (_Float16*)alloc((size_t)512 * 256 * 2);
    _Float16* WtCaV  = (_Float16*)alloc((size_t)512 * 256 * 2);
    _Float16* WtCaOf = (_Float16*)alloc((size_t)96  * 256 * 2);
    _Float16* WtCaO  = (_Float16*)alloc((size_t)256 * 512 * 2);
    _Float16* WtF1   = (_Float16*)alloc((size_t)1024 * 256 * 2);
    _Float16* WtF2   = (_Float16*)alloc((size_t)256 * 1024 * 2);
    float* Qf       = (float*)alloc((size_t)M_Q * 256 * 4);
    float* Kf       = (float*)alloc((size_t)M_Q * 256 * 4);
    float* Vf       = (float*)alloc((size_t)M_Q * 256 * 4);
    float* attn_out = (float*)alloc((size_t)M_Q * 256 * 4);
    float* sa_proj  = (float*)alloc((size_t)M_Q * 256 * 4);
    float* xbuf     = (float*)alloc((size_t)M_Q * 256 * 4);
    float* qhb      = (float*)alloc((size_t)M_Q * 512 * 4);
    float* refb     = (float*)alloc((size_t)M_Q * 3 * 4);
    float* offb     = (float*)alloc((size_t)M_Q * 96 * 4);
    _Float16* kvol  = (_Float16*)alloc((size_t)M_KV * 512 * 2);
    _Float16* vvol  = (_Float16*)alloc((size_t)M_KV * 512 * 2);
    _Float16* fkvol = (_Float16*)alloc((size_t)M_KV * 512 * 2);
    _Float16* fvvol = (_Float16*)alloc((size_t)M_KV * 512 * 2);
    float* sampb    = (float*)alloc((size_t)M_Q * 512 * 4);
    float* ca_proj  = (float*)alloc((size_t)M_Q * 256 * 4);
    float* ybuf     = (float*)alloc((size_t)M_Q * 256 * 4);
    float* h1       = (float*)alloc((size_t)M_Q * 1024 * 4);
    float* h2       = (float*)alloc((size_t)M_Q * 256 * 4);

    auto wt = [&](const float* W, _Float16* Wt, int K, int N) {
        wt_f16_kernel<<<dim3((K * N + 255) / 256), 256, 0, stream>>>(W, Wt, K, N);
    };
    auto gemm = [&](const float* A0, const float* A1, const _Float16* Wt, const float* bias,
                    float* Cf, _Float16* Ch, int M, int N, int K, int act) {
        dim3 grid((N / 16 + 3) / 4, M / 32);
        if (A1)
            gemm_wmma_kernel<true><<<grid, 128, 0, stream>>>(A0, A1, Wt, bias, Cf, Ch, M, N, K, act);
        else
            gemm_wmma_kernel<false><<<grid, 128, 0, stream>>>(A0, A1, Wt, bias, Cf, Ch, M, N, K, act);
    };

    // --- weight transposition to f16 [N,K] ---
    wt(sa_Wq, WtSaQ, 256, 256);   wt(sa_Wk, WtSaK, 256, 256);
    wt(sa_Wv, WtSaV, 256, 256);   wt(sa_Wo, WtSaO, 256, 256);
    wt(ca_Wq, WtCaQ, 256, 512);   wt(ca_Wk, WtCaK, 256, 512);
    wt(ca_Wv, WtCaV, 256, 512);   wt(ca_Woff, WtCaOf, 256, 96);
    wt(ca_Wo, WtCaO, 512, 256);   wt(f_W1, WtF1, 256, 1024);
    wt(f_W2, WtF2, 1024, 256);

    // --- self-attention ---
    gemm(query, qpos, WtSaQ, sa_bq, Qf, nullptr, M_Q, 256, 256, 0);
    gemm(query, qpos, WtSaK, sa_bk, Kf, nullptr, M_Q, 256, 256, 0);
    gemm(query, nullptr, WtSaV, sa_bv, Vf, nullptr, M_Q, 256, 256, 0);
    flash_attn_kernel<<<dim3(NQ / 16, 8, BATCH), 32, 0, stream>>>(Qf, Kf, Vf, attn_out);
    gemm(attn_out, nullptr, WtSaO, sa_bo, sa_proj, nullptr, M_Q, 256, 256, 0);
    add_ln_kernel<<<M_Q, 256, 0, stream>>>(query, sa_proj, ln1_g, ln1_b, xbuf);

    // --- deformable cross-attention ---
    gemm(xbuf, nullptr, WtCaQ, ca_bq, qhb, nullptr, M_Q, 512, 256, 0);
    ref_kernel<<<(M_Q * 3 + 255) / 256, 256, 0, stream>>>(xbuf, ca_Wref, ca_bref, refb);
    gemm(xbuf, nullptr, WtCaOf, ca_boff, offb, nullptr, M_Q, 96, 256, 0);
    gemm(pos, img,  WtCaK, ca_bk, nullptr, kvol,  M_KV, 512, 256, 0);
    gemm(img, nullptr, WtCaV, ca_bv, nullptr, vvol,  M_KV, 512, 256, 0);
    gemm(pos, flow, WtCaK, ca_bk, nullptr, fkvol, M_KV, 512, 256, 0);
    gemm(flow, nullptr, WtCaV, ca_bv, nullptr, fvvol, M_KV, 512, 256, 0);
    deform_sample_kernel<<<M_Q, 256, 0, stream>>>(qhb, refb, offb,
                                                  kvol, vvol, fkvol, fvvol, sampb);
    gemm(sampb, nullptr, WtCaO, ca_bo, ca_proj, nullptr, M_Q, 256, 512, 0);
    add_ln_kernel<<<M_Q, 256, 0, stream>>>(xbuf, ca_proj, ln2_g, ln2_b, ybuf);

    // --- FFN ---
    gemm(ybuf, nullptr, WtF1, f_b1, h1, nullptr, M_Q, 1024, 256, 1);
    gemm(h1, nullptr, WtF2, f_b2, h2, nullptr, M_Q, 256, 1024, 0);
    add_ln_kernel<<<M_Q, 256, 0, stream>>>(ybuf, h2, ln3_g, ln3_b, (float*)d_out);
}
